// ConvSelfAttention_7842610282526
// MI455X (gfx1250) — compile-verified, matrix operand order checked by reading, not run on hardware
//
#include <hip/hip_runtime.h>
#include <hip/hip_bf16.h>

#define B_      4
#define CIN     128
#define LEN     2048
#define HEADS_  8
#define CHEAD   32
#define HIDDEN_ 256

typedef __attribute__((ext_vector_type(16))) __bf16 v16bf;
typedef __attribute__((ext_vector_type(8)))  float  v8f;

union BFrag {
    v16bf v;
    unsigned short u[16];
    uint4 q2[2];
};

__device__ __forceinline__ unsigned short f2bf(float f) {
    union { float f; unsigned int u; } c; c.f = f;
    unsigned int u = c.u;
    u += 0x7FFFu + ((u >> 16) & 1u);   // round-to-nearest-even
    return (unsigned short)(u >> 16);
}

__device__ __forceinline__ v8f wmma_bf16(const BFrag& a, const BFrag& b, v8f c) {
    return __builtin_amdgcn_wmma_f32_16x16x32_bf16(false, a.v, false, b.v,
                                                   (short)0, c, false, false);
}

// ---------------------------------------------------------------------------
// Kernel 1: QKV projection. qkv[b,o,l] = w_qkv[o,:] . x[b,:,l] + b_qkv[o]
// 64(o) x 16(l) per block, 4 waves, K=128 in 4 bf16 WMMA steps.
// q rows pre-scaled by 1/sqrt(L) (reference quirk). q,k stored (b,h,l,c);
// v stored TRANSPOSED (b,h,c,l) so kernel 2b can async-DMA it into LDS.
// oTile is wave-uniform (16-row tile never crosses a 256 boundary) and is
// forced into an SGPR via readfirstlane so the q/k/v store selection is a
// scalar branch, not an EXEC-mask dance.
// ---------------------------------------------------------------------------
__global__ __launch_bounds__(128) void qkv_proj_kernel(
    const float* __restrict__ x, const float* __restrict__ w_qkv,
    const float* __restrict__ b_qkv,
    unsigned short* __restrict__ qws, unsigned short* __restrict__ kws,
    unsigned short* __restrict__ vtws)
{
    __shared__ __align__(16) unsigned short wT[64 * 128];  // [o_local][i] bf16
    __shared__ __align__(16) unsigned short xT[16 * 128];  // [l_local][i] bf16

    const int tid  = threadIdx.x;
    const int lane = tid & 31, wave = tid >> 5;
    const int half = lane >> 4, nm = lane & 15;

    int blk  = blockIdx.x;
    int lBlk = blk % (LEN / 16); blk /= (LEN / 16);
    int oBlk = blk % 12;         blk /= 12;
    int b    = blk;
    const int oBase = oBlk * 64;
    const int lBase = lBlk * 16;

    { // stage 64x128 weight tile as bf16
        int o  = tid >> 1;
        int i0 = (tid & 1) * 64;
        const float* src = w_qkv + (size_t)(oBase + o) * CIN + i0;
        unsigned short* dst = wT + o * 128 + i0;
        #pragma unroll 8
        for (int i = 0; i < 64; ++i) dst[i] = f2bf(src[i]);
    }
    { // stage x tile transposed: xT[l][i]  (B-fragment layout [N][K])
        int l  = tid & 15;
        int i0 = (tid >> 4) * 16;
        unsigned short* dst = xT + l * 128 + i0;
        #pragma unroll
        for (int i = 0; i < 16; ++i)
            dst[i] = f2bf(x[((size_t)b * CIN + (i0 + i)) * LEN + lBase + l]);
    }
    __syncthreads();

    v8f acc = {};
    #pragma unroll
    for (int kc = 0; kc < 4; ++kc) {
        BFrag a, bb;
        const unsigned short* ap = wT + (wave * 16 + nm) * 128 + kc * 32 + half * 8;
        a.q2[0] = *(const uint4*)(ap);
        a.q2[1] = *(const uint4*)(ap + 16);
        const unsigned short* bp = xT + nm * 128 + kc * 32 + half * 8;
        bb.q2[0] = *(const uint4*)(bp);
        bb.q2[1] = *(const uint4*)(bp + 16);
        acc = wmma_bf16(a, bb, acc);
    }

    const float inv_sqrt_L = 0.022097086912079608f;  // 1/sqrt(2048)
    // wave-uniform tile row base; pin to SGPR so sel/h/cBase are scalar
    const int oTile = __builtin_amdgcn_readfirstlane(oBase + wave * 16);
    const int sel   = oTile >> 8;           // 0=q 1=k 2=v   (SGPR)
    const int h     = (oTile >> 5) & 7;     // SGPR
    const int cBase = oTile & 31;           // 0 or 16       (SGPR)
    const int l     = lBase + nm;
    const size_t bh = (size_t)b * HEADS_ + h;

    if (sel == 0) {
        #pragma unroll
        for (int r = 0; r < 8; ++r) {        // C layout: M = r + 8*half
            int m = r + half * 8;
            float v = acc[r] + b_qkv[oTile + m];
            qws[(bh * LEN + l) * CHEAD + cBase + m] = f2bf(v * inv_sqrt_L);
        }
    } else if (sel == 1) {
        #pragma unroll
        for (int r = 0; r < 8; ++r) {
            int m = r + half * 8;
            float v = acc[r] + b_qkv[oTile + m];
            kws[(bh * LEN + l) * CHEAD + cBase + m] = f2bf(v);
        }
    } else {
        #pragma unroll
        for (int r = 0; r < 8; ++r) {
            int m = r + half * 8;
            float v = acc[r] + b_qkv[oTile + m];
            vtws[(bh * CHEAD + cBase + m) * LEN + l] = f2bf(v);   // (b,h,c,l)
        }
    }
}

// ---------------------------------------------------------------------------
// Kernel 2a: per-query softmax stats over keys (online max/sum, flash-style).
// Block = 16 queries; 4 waves stream disjoint 16-key chunks; combine in LDS.
// ---------------------------------------------------------------------------
__global__ __launch_bounds__(128) void softmax_stats_kernel(
    const unsigned short* __restrict__ qws, const unsigned short* __restrict__ kws,
    float* __restrict__ rowMax, float* __restrict__ rowSum)
{
    __shared__ float pm[4][16];
    __shared__ float ps[4][16];

    const int tid  = threadIdx.x;
    const int lane = tid & 31, wave = tid >> 5;
    const int half = lane >> 4, nm = lane & 15;

    int blk  = blockIdx.x;
    int qBlk = blk % (LEN / 16); blk /= (LEN / 16);
    int h    = blk % HEADS_;     blk /= HEADS_;
    int b    = blk;
    const int qBase = qBlk * 16;
    const size_t headOff = ((size_t)b * HEADS_ + h) * LEN;

    BFrag a;  // A fragment: 16 query rows, loop invariant
    {
        const unsigned short* ap = qws + (headOff + qBase + nm) * CHEAD + half * 8;
        a.q2[0] = *(const uint4*)ap;
        a.q2[1] = *(const uint4*)(ap + 16);
    }

    float rMax[8], rSum[8];
    #pragma unroll
    for (int r = 0; r < 8; ++r) { rMax[r] = -3.0e38f; rSum[r] = 0.f; }

    for (int ch = 0; ch < LEN / 64; ++ch) {
        int kBase = (ch * 4 + wave) * 16;
        const unsigned short* bp = kws + (headOff + kBase + nm) * CHEAD + half * 8;
        BFrag bb;
        bb.q2[0] = *(const uint4*)bp;
        bb.q2[1] = *(const uint4*)(bp + 16);
        __builtin_prefetch(bp + 2048, 0, 3);   // next 64-key chunk (L2 resident)

        v8f s = {};
        s = wmma_bf16(a, bb, s);

        #pragma unroll
        for (int r = 0; r < 8; ++r) {
            float cm = s[r];
            cm = fmaxf(cm, __shfl_xor(cm, 1, 32));
            cm = fmaxf(cm, __shfl_xor(cm, 2, 32));
            cm = fmaxf(cm, __shfl_xor(cm, 4, 32));
            cm = fmaxf(cm, __shfl_xor(cm, 8, 32));
            float nMax = fmaxf(rMax[r], cm);
            float p = __expf(s[r] - nMax);
            float cs = p;
            cs += __shfl_xor(cs, 1, 32);
            cs += __shfl_xor(cs, 2, 32);
            cs += __shfl_xor(cs, 4, 32);
            cs += __shfl_xor(cs, 8, 32);
            rSum[r] = rSum[r] * __expf(rMax[r] - nMax) + cs;
            rMax[r] = nMax;
        }
    }

    if (nm == 0) {  // lane 0 holds rows 0-7, lane 16 rows 8-15 (reduced)
        #pragma unroll
        for (int r = 0; r < 8; ++r) {
            pm[wave][half * 8 + r] = rMax[r];
            ps[wave][half * 8 + r] = rSum[r];
        }
    }
    __syncthreads();
    if (tid < 16) {
        float m = pm[0][tid];
        m = fmaxf(m, pm[1][tid]); m = fmaxf(m, pm[2][tid]); m = fmaxf(m, pm[3][tid]);
        float ssum = 0.f;
        #pragma unroll
        for (int w2 = 0; w2 < 4; ++w2) ssum += ps[w2][tid] * __expf(pm[w2][tid] - m);
        rowMax[headOff + qBase + tid] = m;
        rowSum[headOff + qBase + tid] = ssum;
    }
}

// ---------------------------------------------------------------------------
// Kernel 2b: attn[a,d] = sum_q P[q,a] * v[q,d]  (reference contracts over q).
// Per wave: 16 a's x 32 d's; loop over q in chunks of 32 (= WMMA K).
// V chunk (already transposed in ws) is DMA'd LDS-ward with
// GLOBAL_LOAD_ASYNC_TO_LDS_B128, overlapped with the S-tile WMMA + exp work,
// fenced with s_wait_asynccnt before the block barrier.
// ---------------------------------------------------------------------------
__global__ __launch_bounds__(128) void attn_pv_kernel(
    const unsigned short* __restrict__ qws, const unsigned short* __restrict__ kws,
    const unsigned short* __restrict__ vtws,
    const float* __restrict__ rowMax, const float* __restrict__ rowSum,
    unsigned short* __restrict__ attnws)
{
    __shared__ __align__(16) unsigned short pT[4][16 * 32]; // per-wave P^T [a][q]
    __shared__ __align__(16) unsigned short vT[32 * 32];    // V^T [d][q] shared

    const int tid  = threadIdx.x;
    const int lane = tid & 31, wave = tid >> 5;
    const int half = lane >> 4, nm = lane & 15;

    int blk  = blockIdx.x;
    int aBlk = blk % (LEN / 64); blk /= (LEN / 64);
    int h    = blk % HEADS_;     blk /= HEADS_;
    int b    = blk;
    const int aBase = aBlk * 64 + wave * 16;
    const size_t headOff = ((size_t)b * HEADS_ + h) * LEN;

    // Per-thread async-copy slot: thread t moves 16B: row d = t>>2, q0 = (t&3)*8
    const int vd  = tid >> 2;
    const int vq0 = (tid & 3) * 8;
    const unsigned vLds = (unsigned)(uintptr_t)(void*)&vT[vd * 32 + vq0];
    const unsigned short* vSrcRow =
        vtws + (((size_t)b * HEADS_ + h) * CHEAD + vd) * LEN + vq0;

    BFrag kb;  // B fragment for S: the 16 k-rows 'a' of this wave (loop invariant)
    {
        const unsigned short* kp = kws + (headOff + aBase + nm) * CHEAD + half * 8;
        kb.q2[0] = *(const uint4*)kp;
        kb.q2[1] = *(const uint4*)(kp + 16);
    }

    v8f acc0 = {}, acc1 = {};

    for (int qc = 0; qc < LEN / 32; ++qc) {
        const int qBase = qc * 32;

        // kick off async V-chunk DMA to LDS (ASYNCcnt), overlap with S/P math
        asm volatile("global_load_async_to_lds_b128 %0, %1, off"
                     :: "v"(vLds), "v"(vSrcRow + qBase) : "memory");

        #pragma unroll
        for (int t = 0; t < 2; ++t) {  // two 16-query S tiles
            const unsigned short* qp =
                qws + (headOff + qBase + t * 16 + nm) * CHEAD + half * 8;
            BFrag qa;
            qa.q2[0] = *(const uint4*)qp;
            qa.q2[1] = *(const uint4*)(qp + 16);
            v8f s = {};
            s = wmma_bf16(qa, kb, s);
            #pragma unroll
            for (int r = 0; r < 8; ++r) {      // C layout: M=q'=r+8*half, N=a=nm
                int q = qBase + t * 16 + r + half * 8;
                float m  = rowMax[headOff + q];
                float sm = rowSum[headOff + q];
                float p  = __expf(s[r] - m) / sm;
                pT[wave][nm * 32 + t * 16 + r + half * 8] = f2bf(p);
            }
        }

        asm volatile("s_wait_asynccnt 0x0" ::: "memory");
        __syncthreads();

        BFrag pa;  // A fragment from P^T: M=a, K=q
        const unsigned short* pp = &pT[wave][nm * 32 + half * 8];
        pa.q2[0] = *(const uint4*)pp;
        pa.q2[1] = *(const uint4*)(pp + 16);

        BFrag vb0, vb1;  // B fragments from V^T: N=d (0..15 / 16..31), K=q
        const unsigned short* vp0 = vT + nm * 32 + half * 8;
        vb0.q2[0] = *(const uint4*)vp0;
        vb0.q2[1] = *(const uint4*)(vp0 + 16);
        const unsigned short* vp1 = vT + (16 + nm) * 32 + half * 8;
        vb1.q2[0] = *(const uint4*)vp1;
        vb1.q2[1] = *(const uint4*)(vp1 + 16);

        acc0 = wmma_bf16(pa, vb0, acc0);
        acc1 = wmma_bf16(pa, vb1, acc1);
        __syncthreads();
    }

    // attn workspace layout [b][l=a][o=h*32+d] bf16 (B-fragment friendly for k3)
    #pragma unroll
    for (int r = 0; r < 8; ++r) {
        int aa = aBase + r + half * 8;
        size_t rowOff = ((size_t)b * LEN + aa) * HIDDEN_ + h * CHEAD;
        attnws[rowOff + nm]      = f2bf(acc0[r]);
        attnws[rowOff + 16 + nm] = f2bf(acc1[r]);
    }
}

// ---------------------------------------------------------------------------
// Kernel 3: out = BN( w_out @ attn + b_out + x ). M=128, N=2048, K=256.
// ---------------------------------------------------------------------------
__global__ __launch_bounds__(128) void out_proj_bn_kernel(
    const float* __restrict__ x, const float* __restrict__ w_out,
    const float* __restrict__ b_out,
    const float* __restrict__ bn_w, const float* __restrict__ bn_b,
    const float* __restrict__ bn_m, const float* __restrict__ bn_v,
    const unsigned short* __restrict__ attnws, float* __restrict__ out)
{
    const int tid  = threadIdx.x;
    const int lane = tid & 31, wave = tid >> 5;
    const int half = lane >> 4, nm = lane & 15;

    int blk  = blockIdx.x;
    int lBlk = blk % (LEN / 16); blk /= (LEN / 16);
    int oBlk = blk % 2;          blk /= 2;
    int b    = blk;
    const int oBase = oBlk * 64 + wave * 16;
    const int lBase = lBlk * 16;
    const int oRow  = oBase + nm;
    const int lCol  = lBase + nm;

    v8f acc = {};
    #pragma unroll
    for (int kc = 0; kc < 8; ++kc) {
        BFrag a;  // on-the-fly f32 -> bf16 of w_out row (small, L2 hot)
        const float* ap = w_out + (size_t)oRow * HIDDEN_ + kc * 32 + half * 8;
        #pragma unroll
        for (int j = 0; j < 8; ++j) a.u[j] = f2bf(ap[j]);
        #pragma unroll
        for (int j = 0; j < 8; ++j) a.u[8 + j] = f2bf(ap[16 + j]);

        BFrag bb;
        const unsigned short* bp =
            attnws + ((size_t)b * LEN + lCol) * HIDDEN_ + kc * 32 + half * 8;
        bb.q2[0] = *(const uint4*)bp;
        bb.q2[1] = *(const uint4*)(bp + 16);
        acc = wmma_bf16(a, bb, acc);
    }

    #pragma unroll
    for (int r = 0; r < 8; ++r) {
        int o = oBase + r + half * 8;
        float inv = bn_w[o] * rsqrtf(bn_v[o] + 1e-5f);
        float add = bn_b[o] - bn_m[o] * inv;
        size_t xi = ((size_t)b * CIN + o) * LEN + lCol;
        float v = acc[r] + b_out[o] + x[xi];
        out[xi] = v * inv + add;
    }
}

// ---------------------------------------------------------------------------
extern "C" void kernel_launch(void* const* d_in, const int* in_sizes, int n_in,
                              void* d_out, int out_size, void* d_ws, size_t ws_size,
                              hipStream_t stream) {
    const float* x     = (const float*)d_in[0];
    const float* w_qkv = (const float*)d_in[1];
    const float* b_qkv = (const float*)d_in[2];
    const float* w_out = (const float*)d_in[3];
    const float* b_out = (const float*)d_in[4];
    const float* bn_w  = (const float*)d_in[5];
    const float* bn_b  = (const float*)d_in[6];
    const float* bn_m  = (const float*)d_in[7];
    const float* bn_v  = (const float*)d_in[8];

    char* ws = (char*)d_ws;
    const size_t qkvElems = (size_t)B_ * HEADS_ * LEN * CHEAD;   // 2M bf16 each
    unsigned short* qws    = (unsigned short*)ws; ws += qkvElems * 2;
    unsigned short* kws    = (unsigned short*)ws; ws += qkvElems * 2;
    unsigned short* vtws   = (unsigned short*)ws; ws += qkvElems * 2;
    unsigned short* attnws = (unsigned short*)ws; ws += (size_t)B_ * LEN * HIDDEN_ * 2;
    float* rowMax = (float*)ws; ws += (size_t)B_ * HEADS_ * LEN * 4;
    float* rowSum = (float*)ws;

    qkv_proj_kernel<<<dim3(B_ * 12 * (LEN / 16)), 128, 0, stream>>>(
        x, w_qkv, b_qkv, qws, kws, vtws);
    softmax_stats_kernel<<<dim3(B_ * HEADS_ * (LEN / 16)), 128, 0, stream>>>(
        qws, kws, rowMax, rowSum);
    attn_pv_kernel<<<dim3(B_ * HEADS_ * (LEN / 64)), 128, 0, stream>>>(
        qws, kws, vtws, rowMax, rowSum, attnws);
    out_proj_bn_kernel<<<dim3(B_ * 2 * (LEN / 16)), 128, 0, stream>>>(
        x, w_out, b_out, bn_w, bn_b, bn_m, bn_v, attnws, (float*)d_out);
}
